// LSTM_88330297409630
// MI455X (gfx1250) — compile-verified, matrix operand order checked by reading, not run on hardware
//
#include <hip/hip_runtime.h>

// ---------------------------------------------------------------------------
// LSTM acceptor on MI455X (gfx1250, wave32, WMMA).
// gates = [x_t | h] @ Wcat^T + (b_ih + b_hh), fused with the cell update.
// Wcat (bf16, 12 MB) stays resident in the 192 MB L2; one kernel per step.
// K-loop is software-pipelined: next A/B fragments are loaded while the
// current 4 WMMAs execute, so loads overlap the matrix pipe.
// ---------------------------------------------------------------------------

typedef __attribute__((ext_vector_type(16))) __bf16 v16bf;
typedef __attribute__((ext_vector_type(8)))  float  v8f;

constexpr int B = 128, S = 1024, I = 512, H = 1024;
constexpr int K  = I + H;     // 1536 : concat(x, h) width
constexpr int G4 = 4 * H;     // 4096 : i,f,g,o gate columns

// workspace layout (bytes) -- total ~13.3 MB
constexpr size_t WCAT_OFF  = 0;                               // bf16 [4H][K]
constexpr size_t ZBUF0_OFF = (size_t)G4 * K * 2;              // bf16 [B][K]
constexpr size_t ZBUF1_OFF = ZBUF0_OFF + (size_t)B * K * 2;   // bf16 [B][K]
constexpr size_t C_OFF     = ZBUF1_OFF + (size_t)B * K * 2;   // f32  [B][H]
constexpr size_t BSUM_OFF  = C_OFF + (size_t)B * H * 4;       // f32  [4H]

__device__ __forceinline__ __bf16 to_bf16(float f) {
  // round-to-nearest-even f32 -> bf16
  unsigned u = __builtin_bit_cast(unsigned, f);
  unsigned r = u + 0x7FFFu + ((u >> 16) & 1u);
  return __builtin_bit_cast(__bf16, (unsigned short)(r >> 16));
}

__device__ __forceinline__ float sigmoidf_dev(float x) {
  return 1.0f / (1.0f + __expf(-x));
}

// --- one-time: Wcat[row][col] = bf16(col<I ? W_ih : W_hh), bsum = b_ih + b_hh
__global__ void prep_weights(const float* __restrict__ Wih,
                             const float* __restrict__ Whh,
                             const float* __restrict__ bih,
                             const float* __restrict__ bhh,
                             __bf16* __restrict__ Wcat,
                             float*  __restrict__ bsum) {
  int idx = blockIdx.x * blockDim.x + threadIdx.x;
  if (idx < G4 * K) {
    int row = idx / K, col = idx % K;
    float v = (col < I) ? Wih[(size_t)row * I + col]
                        : Whh[(size_t)row * H + (col - I)];
    Wcat[idx] = to_bf16(v);
  }
  if (idx < G4) bsum[idx] = bih[idx] + bhh[idx];
}

// --- one-time: z0 = [bf16(x_0) | 0], c = 0
__global__ void init_state(const float* __restrict__ seq,
                           __bf16* __restrict__ z0,
                           float*  __restrict__ c) {
  int idx = blockIdx.x * blockDim.x + threadIdx.x;
  if (idx < B * K) {
    int b = idx / K, j = idx % K;
    float v = (j < I) ? seq[(size_t)b * S * I + j] : 0.0f;
    z0[idx] = to_bf16(v);
  }
  int ci = idx - B * K;
  if (ci >= 0 && ci < B * H) c[ci] = 0.0f;
}

#define WMMA_BF16(A, Bm, C) \
  __builtin_amdgcn_wmma_f32_16x16x32_bf16(false, (A), false, (Bm), (short)0, (C), false, false)

// --- per-step fused GEMM + LSTM cell update.
// grid = 8 batch-tiles x 16 col-groups = 128 blocks, 128 threads (4 waves).
// wave w computes gate w's 16x64 slab; after the barrier the block does the
// pointwise update for its 16(batch) x 64(hidden) patch.
__global__ __launch_bounds__(128)
void lstm_step(const __bf16* __restrict__ zcur,   // [B][K] bf16, [x_t | h_{t-1}]
               __bf16*       __restrict__ znext,  // [B][K] bf16, gets h_t and x_{t+1}
               const __bf16* __restrict__ Wcat,   // [4H][K] bf16
               float*        __restrict__ c,      // [B][H] f32 cell state
               const float*  __restrict__ bsum,   // [4H] f32, b_ih + b_hh
               const float*  __restrict__ seq,    // [B][S][I] f32
               const int*    __restrict__ len,    // [B]
               float*        __restrict__ out,    // [B][H] f32
               int t) {
  __shared__ float ldsG[4][16][64];               // 16 KB of 320 KB WGP LDS

  const int tid   = threadIdx.x;
  const int lane  = tid & 31;
  const int wave  = tid >> 5;        // gate id: 0=i 1=f 2=g 3=o
  const int bt    = blockIdx.x >> 4; // batch tile 0..7
  const int ng    = blockIdx.x & 15; // 64-wide hidden col group 0..15
  const int l16   = lane & 15;
  const int khalf = (lane >> 4) * 16;

  // Stage x_{t+1} first: writes only the next z-buffer (unread this step),
  // so these loads/stores overlap the GEMM below.
  if (t + 1 < S) {
    int gid = blockIdx.x * 128 + tid;
    for (int p = gid; p < B * I; p += 128 * 128) {
      int b = p >> 9;            // / I (I = 512)
      int i = p & (I - 1);
      znext[(size_t)b * K + i] =
          to_bf16(seq[((size_t)b * S + (t + 1)) * I + i]);
    }
  }

  // Fragment layouts (this code's k order for 16x32 A / 32x16 B, same in both; not the ISA table order): lane L holds 16
  // contiguous K-values of row/col (L&15) starting at K=(L>>4)*16 -> plain
  // 32-byte contiguous per-lane loads.
  const __bf16* aptr = zcur + (size_t)(bt * 16 + l16) * K + khalf;
  const __bf16* bptr = Wcat + (size_t)(wave * H + ng * 64 + l16) * K + khalf;

  v8f acc0 = {}, acc1 = {}, acc2 = {}, acc3 = {};

  // software-pipelined K loop: load k+32 fragments while k's WMMAs execute
  v16bf a  = *(const v16bf*)(aptr);
  v16bf b0 = *(const v16bf*)(bptr + (size_t) 0 * K);
  v16bf b1 = *(const v16bf*)(bptr + (size_t)16 * K);
  v16bf b2 = *(const v16bf*)(bptr + (size_t)32 * K);
  v16bf b3 = *(const v16bf*)(bptr + (size_t)48 * K);
#pragma unroll 2
  for (int k = 32; k < K; k += 32) {
    v16bf an  = *(const v16bf*)(aptr + k);
    v16bf b0n = *(const v16bf*)(bptr + (size_t) 0 * K + k);
    v16bf b1n = *(const v16bf*)(bptr + (size_t)16 * K + k);
    v16bf b2n = *(const v16bf*)(bptr + (size_t)32 * K + k);
    v16bf b3n = *(const v16bf*)(bptr + (size_t)48 * K + k);
    acc0 = WMMA_BF16(a, b0, acc0);
    acc1 = WMMA_BF16(a, b1, acc1);
    acc2 = WMMA_BF16(a, b2, acc2);
    acc3 = WMMA_BF16(a, b3, acc3);
    a = an; b0 = b0n; b1 = b1n; b2 = b2n; b3 = b3n;
  }
  acc0 = WMMA_BF16(a, b0, acc0);
  acc1 = WMMA_BF16(a, b1, acc1);
  acc2 = WMMA_BF16(a, b2, acc2);
  acc3 = WMMA_BF16(a, b3, acc3);

  // C/D layout: VGPR r, lane L -> row r + 8*(L>>4), col L&15
  const int rrow = 8 * (lane >> 4);
#pragma unroll
  for (int r = 0; r < 8; ++r) {
    ldsG[wave][rrow + r][ 0 + l16] = acc0[r];
    ldsG[wave][rrow + r][16 + l16] = acc1[r];
    ldsG[wave][rrow + r][32 + l16] = acc2[r];
    ldsG[wave][rrow + r][48 + l16] = acc3[r];
  }
  __syncthreads();

  // pointwise LSTM update over the 16x64 patch (1024 elems / 128 threads)
  for (int e = tid; e < 16 * 64; e += 128) {
    int row = e >> 6, col = e & 63;
    int b = bt * 16 + row;
    int n = ng * 64 + col;
    float gi = ldsG[0][row][col] + bsum[0 * H + n];
    float gf = ldsG[1][row][col] + bsum[1 * H + n];
    float gg = ldsG[2][row][col] + bsum[2 * H + n];
    float go = ldsG[3][row][col] + bsum[3 * H + n];
    float ig = sigmoidf_dev(gi);
    float fg = sigmoidf_dev(gf);
    float og = sigmoidf_dev(go);
    float gt = tanhf(gg);
    size_t ci = (size_t)b * H + n;
    float cn = fg * c[ci] + ig * gt;
    c[ci] = cn;
    float h = og * tanhf(cn);
    znext[(size_t)b * K + I + n] = to_bf16(h);
    if (t == len[b] - 1) out[ci] = h;   // capture h at the last valid step
  }
}

extern "C" void kernel_launch(void* const* d_in, const int* in_sizes, int n_in,
                              void* d_out, int out_size, void* d_ws, size_t ws_size,
                              hipStream_t stream) {
  (void)in_sizes; (void)n_in; (void)out_size; (void)ws_size;
  const float* seq = (const float*)d_in[0];
  const int*   len = (const int*)  d_in[1];
  const float* Wih = (const float*)d_in[2];
  const float* Whh = (const float*)d_in[3];
  const float* bih = (const float*)d_in[4];
  const float* bhh = (const float*)d_in[5];
  float* out = (float*)d_out;

  char* ws = (char*)d_ws;
  __bf16* Wcat = (__bf16*)(ws + WCAT_OFF);
  __bf16* z0   = (__bf16*)(ws + ZBUF0_OFF);
  __bf16* z1   = (__bf16*)(ws + ZBUF1_OFF);
  float*  cst  = (float*) (ws + C_OFF);
  float*  bsum = (float*) (ws + BSUM_OFF);

  { int n = G4 * K;        prep_weights<<<(n + 255) / 256, 256, 0, stream>>>(Wih, Whh, bih, bhh, Wcat, bsum); }
  { int n = B * K + B * H; init_state  <<<(n + 255) / 256, 256, 0, stream>>>(seq, z0, cst); }

  for (int t = 0; t < S; ++t) {
    __bf16* zc = (t & 1) ? z1 : z0;
    __bf16* zn = (t & 1) ? z0 : z1;
    lstm_step<<<128, 128, 0, stream>>>(zc, zn, Wcat, cst, bsum, seq, len, out, t);
  }
}
